// _ToyExpertsLinearLoop_50105088475309
// MI455X (gfx1250) — matrix operand order repaired and checked by reading, not compiled, not run on hardware
//
#include <hip/hip_runtime.h>

// Problem constants (from reference): E=8, T=2048, H=1024, I=4096
#define E_ 8
#define T_ 2048
#define H_ 1024
#define I_ 4096
#define O_ (2 * I_) // 8192 rows of w_gate_up per expert

typedef _Float16 h4   __attribute__((ext_vector_type(4)));
typedef _Float16 h8   __attribute__((ext_vector_type(8)));
typedef _Float16 v16h __attribute__((ext_vector_type(16)));
typedef float    v8f  __attribute__((ext_vector_type(8)));

// Load a 16-halfword WMMA fragment from two contiguous 16-byte chunks at
// base + off and base + off + 16 elements. Offsets are 32-bit so the backend
// can use saddr(base) + voffset + imm (GVS) addressing.
__device__ __forceinline__ v16h load_frag(const _Float16* __restrict__ base,
                                          unsigned off) {
  h8 a = *reinterpret_cast<const h8*>(base + off);
  h8 b = *reinterpret_cast<const h8*>(base + off + 16);
  return __builtin_shufflevector(a, b, 0, 1, 2, 3, 4, 5, 6, 7,
                                 8, 9, 10, 11, 12, 13, 14, 15);
}
// Variant for B fragments: 16 k-strided halves, rows s (s<8) and s+8 (s>=8) from off.
__device__ __forceinline__ v16h load_fragB(const _Float16* __restrict__ base,
                                           unsigned off, unsigned ld) {
  v16h r;
#pragma unroll
  for (int s = 0; s < 16; ++s)
    r[s] = base[off + (unsigned)(s < 8 ? s : s + 8) * ld];
  return r;
}

__device__ __forceinline__ v8f wmma_f16(v16h a, v16h b, v8f c) {
  // D = A*B + C, f16 inputs, f32 accumulate
  return __builtin_amdgcn_wmma_f32_16x16x32_f16(false, a, false, b,
                                                (short)0, c, false, false);
}

// ---------------------------------------------------------------------------
// Pass 0a: straight fp32 -> (hi f16, lo f16) split, 4-wide
// ---------------------------------------------------------------------------
__global__ __launch_bounds__(256) void k_split_f32(const float* __restrict__ in,
                                                   _Float16* __restrict__ oh,
                                                   _Float16* __restrict__ ol,
                                                   int n4) {
  int i = blockIdx.x * 256 + threadIdx.x;
  if (i >= n4) return;
  float4 v = reinterpret_cast<const float4*>(in)[i];
  float vv[4] = {v.x, v.y, v.z, v.w};
  h4 hi, lo;
#pragma unroll
  for (int j = 0; j < 4; ++j) {
    _Float16 h = (_Float16)vv[j];
    hi[j] = h;
    lo[j] = (_Float16)(vv[j] - (float)h);
  }
  reinterpret_cast<h4*>(oh)[i] = hi;
  reinterpret_cast<h4*>(ol)[i] = lo;
}

// ---------------------------------------------------------------------------
// Pass 0b: per-expert tiled transpose (R x C -> C x R) + hi/lo split.
// LDS 32x33 tile kills bank conflicts; both read and write are coalesced.
// ---------------------------------------------------------------------------
__global__ __launch_bounds__(256) void k_transpose_split(const float* __restrict__ in,
                                                         _Float16* __restrict__ oh,
                                                         _Float16* __restrict__ ol,
                                                         int R, int C) {
  __shared__ float tile[32][33];
  const int e = blockIdx.z;
  const size_t base = (size_t)e * R * C;
  const int c0 = blockIdx.x * 32;
  const int r0 = blockIdx.y * 32;
  const int tx = threadIdx.x, ty = threadIdx.y;
#pragma unroll
  for (int j = 0; j < 4; ++j) {
    int y = ty + j * 8;
    tile[y][tx] = in[base + (size_t)(r0 + y) * C + (c0 + tx)];
  }
  __syncthreads();
#pragma unroll
  for (int j = 0; j < 4; ++j) {
    int row = ty + j * 8;
    float v = tile[tx][row];
    _Float16 h = (_Float16)v;
    _Float16 l = (_Float16)(v - (float)h);
    size_t o = base + (size_t)(c0 + row) * R + (r0 + tx);
    oh[o] = h;
    ol[o] = l;
  }
}

// ---------------------------------------------------------------------------
// GEMM1 (fused): per wave, 64x32 tile of act[e] = silu(x.WguT[:,0:I]) * x.WguT[:,I:2I]
// A = x rows (K=H contiguous), B = pre-transposed Wgu (H x 8192, N contiguous).
// Split-f16: 3 WMMAs (hh + lh + hl) per tile per K=32 step.
// All loads: uniform SGPR base + 32-bit per-lane offset + constant imm.
// ---------------------------------------------------------------------------
__global__ __launch_bounds__(32) void k_gemm1(const _Float16* __restrict__ xh,
                                              const _Float16* __restrict__ xl,
                                              const _Float16* __restrict__ guh,
                                              const _Float16* __restrict__ gul,
                                              _Float16* __restrict__ acth,
                                              _Float16* __restrict__ actl) {
  const int e = blockIdx.z;
  const int t0 = blockIdx.y * 64;
  const int i0 = blockIdx.x * 32;
  const int lane = threadIdx.x;
  const int lhalf = lane >> 4;  // 0 or 1
  const int lmod = lane & 15;
  const int ac0 = lhalf * 8;    // A-fragment K chunk base: lanes>=16 start at K=8

  const _Float16* __restrict__ xhe = xh + (size_t)e * T_ * H_;
  const _Float16* __restrict__ xle = xl + (size_t)e * T_ * H_;
  const _Float16* __restrict__ guhe = guh + (size_t)e * H_ * O_;
  const _Float16* __restrict__ gule = gul + (size_t)e * H_ * O_;

  v8f acc[4][2][2];  // [m-tile][n-tile][gate/up]
#pragma unroll
  for (int mt = 0; mt < 4; ++mt)
#pragma unroll
    for (int nt = 0; nt < 2; ++nt)
#pragma unroll
      for (int p = 0; p < 2; ++p) {
        v8f z = {0.f, 0.f, 0.f, 0.f, 0.f, 0.f, 0.f, 0.f};
        acc[mt][nt][p] = z;
      }

  // 32-bit induction offsets (element units)
  unsigned aoff = (unsigned)(t0 + lmod) * H_ + ac0;  // +32 per K-step
  unsigned boff = (unsigned)(lhalf * 8) * O_ + i0 + lmod;  // +32*O_ per K-step

  for (int k0 = 0; k0 < H_; k0 += 32) {
    v16h ah[4], al[4];
#pragma unroll
    for (int mt = 0; mt < 4; ++mt) {
      ah[mt] = load_frag(xhe, aoff + (unsigned)(mt * 16 * H_));
      al[mt] = load_frag(xle, aoff + (unsigned)(mt * 16 * H_));
    }
    v16h bh[2][2], bl[2][2];
#pragma unroll
    for (int nt = 0; nt < 2; ++nt) {
      bh[nt][0] = load_fragB(guhe, boff + (unsigned)(nt * 16), O_);        // gate
      bh[nt][1] = load_fragB(guhe, boff + (unsigned)(I_ + nt * 16), O_);   // up
      bl[nt][0] = load_fragB(gule, boff + (unsigned)(nt * 16), O_);
      bl[nt][1] = load_fragB(gule, boff + (unsigned)(I_ + nt * 16), O_);
    }
#pragma unroll
    for (int mt = 0; mt < 4; ++mt)
#pragma unroll
      for (int nt = 0; nt < 2; ++nt)
#pragma unroll
        for (int p = 0; p < 2; ++p) {
          acc[mt][nt][p] = wmma_f16(ah[mt], bh[nt][p], acc[mt][nt][p]);
          acc[mt][nt][p] = wmma_f16(al[mt], bh[nt][p], acc[mt][nt][p]);
          acc[mt][nt][p] = wmma_f16(ah[mt], bl[nt][p], acc[mt][nt][p]);
        }
    aoff += 32;
    boff += 32u * O_;
  }

  // Fused epilogue: act = silu(gate) * up, re-split to hi/lo f16 for GEMM2.
  _Float16* __restrict__ acthe = acth + (size_t)e * T_ * I_;
  _Float16* __restrict__ actle = actl + (size_t)e * T_ * I_;
#pragma unroll
  for (int mt = 0; mt < 4; ++mt)
#pragma unroll
    for (int nt = 0; nt < 2; ++nt) {
      v8f g = acc[mt][nt][0];
      v8f u = acc[mt][nt][1];
#pragma unroll
      for (int r = 0; r < 8; ++r) {
        float gv = g[r], uv = u[r];
        float a = (gv / (1.0f + __expf(-gv))) * uv;
        _Float16 h = (_Float16)a;
        _Float16 l = (_Float16)(a - (float)h);
        const unsigned row = (unsigned)(t0 + mt * 16 + lhalf * 8 + r);
        const unsigned col = (unsigned)(i0 + nt * 16 + lmod);
        const unsigned idx = row * (unsigned)I_ + col;
        acthe[idx] = h;
        actle[idx] = l;
      }
    }
}

// ---------------------------------------------------------------------------
// GEMM2: out[e,t,h] = act[e,t,:] . WdT[e,:,h]   (K = I = 4096), f32 output.
// Register blocking: 4 m-tiles x 2 n-tiles = 64x32 output per wave.
// ---------------------------------------------------------------------------
__global__ __launch_bounds__(32) void k_gemm2(const _Float16* __restrict__ acth,
                                              const _Float16* __restrict__ actl,
                                              const _Float16* __restrict__ wdh,
                                              const _Float16* __restrict__ wdl,
                                              float* __restrict__ out) {
  const int e = blockIdx.z;
  const int t0 = blockIdx.y * 64;
  const int h0 = blockIdx.x * 32;
  const int lane = threadIdx.x;
  const int lhalf = lane >> 4;
  const int lmod = lane & 15;
  const int ac0 = lhalf * 8;

  const _Float16* __restrict__ ahe = acth + (size_t)e * T_ * I_;
  const _Float16* __restrict__ ale = actl + (size_t)e * T_ * I_;
  const _Float16* __restrict__ bhe = wdh + (size_t)e * I_ * H_;
  const _Float16* __restrict__ ble = wdl + (size_t)e * I_ * H_;

  v8f acc[4][2];
#pragma unroll
  for (int mt = 0; mt < 4; ++mt)
#pragma unroll
    for (int nt = 0; nt < 2; ++nt) {
      v8f z = {0.f, 0.f, 0.f, 0.f, 0.f, 0.f, 0.f, 0.f};
      acc[mt][nt] = z;
    }

  unsigned aoff = (unsigned)(t0 + lmod) * I_ + ac0;  // +32 per K-step
  unsigned boff = (unsigned)(lhalf * 8) * H_ + h0 + lmod;  // +32*H_ per K-step

  for (int k0 = 0; k0 < I_; k0 += 32) {
    v16h ah[4], al[4];
#pragma unroll
    for (int mt = 0; mt < 4; ++mt) {
      ah[mt] = load_frag(ahe, aoff + (unsigned)(mt * 16 * I_));
      al[mt] = load_frag(ale, aoff + (unsigned)(mt * 16 * I_));
    }
    v16h bh[2], bl[2];
#pragma unroll
    for (int nt = 0; nt < 2; ++nt) {
      bh[nt] = load_fragB(bhe, boff + (unsigned)(nt * 16), H_);
      bl[nt] = load_fragB(ble, boff + (unsigned)(nt * 16), H_);
    }
#pragma unroll
    for (int mt = 0; mt < 4; ++mt)
#pragma unroll
      for (int nt = 0; nt < 2; ++nt) {
        acc[mt][nt] = wmma_f16(ah[mt], bh[nt], acc[mt][nt]);
        acc[mt][nt] = wmma_f16(al[mt], bh[nt], acc[mt][nt]);
        acc[mt][nt] = wmma_f16(ah[mt], bl[nt], acc[mt][nt]);
      }
    aoff += 32;
    boff += 32u * H_;
  }

  float* __restrict__ oute = out + (size_t)e * T_ * H_;
#pragma unroll
  for (int mt = 0; mt < 4; ++mt)
#pragma unroll
    for (int nt = 0; nt < 2; ++nt) {
      v8f c = acc[mt][nt];
#pragma unroll
      for (int r = 0; r < 8; ++r) {
        const unsigned row = (unsigned)(t0 + mt * 16 + lhalf * 8 + r);
        const unsigned col = (unsigned)(h0 + nt * 16 + lmod);
        oute[row * (unsigned)H_ + col] = c[r];
      }
    }
}

// ---------------------------------------------------------------------------
// Launch: split/transpose pre-passes, then two WMMA GEMMs, all on `stream`.
// Workspace layout (f16 elements):
//   xh,xl: E*T*H each | guT hi,lo: E*H*2I each | wdT hi,lo: E*I*H each |
//   act hi,lo: E*T*I each  => ~704 MB total.
// ---------------------------------------------------------------------------
extern "C" void kernel_launch(void* const* d_in, const int* in_sizes, int n_in,
                              void* d_out, int out_size, void* d_ws, size_t ws_size,
                              hipStream_t stream) {
  (void)in_sizes; (void)n_in; (void)out_size; (void)ws_size;
  const float* x   = (const float*)d_in[0];
  const float* wgu = (const float*)d_in[1];
  const float* wd  = (const float*)d_in[2];
  float* out = (float*)d_out;

  const size_t XN   = (size_t)E_ * T_ * H_;   // 16,777,216
  const size_t GUN  = (size_t)E_ * H_ * O_;   // 67,108,864
  const size_t WDN  = (size_t)E_ * I_ * H_;   // 33,554,432
  const size_t ACTN = (size_t)E_ * T_ * I_;   // 67,108,864

  _Float16* p = (_Float16*)d_ws;
  _Float16* xh  = p; p += XN;
  _Float16* xl  = p; p += XN;
  _Float16* guh = p; p += GUN;
  _Float16* gul = p; p += GUN;
  _Float16* wdh = p; p += WDN;
  _Float16* wdl = p; p += WDN;
  _Float16* acth = p; p += ACTN;
  _Float16* actl = p; p += ACTN;

  // Pass 0: precision split + weight transposes
  {
    const int n4 = (int)(XN / 4);
    k_split_f32<<<dim3((n4 + 255) / 256), dim3(256), 0, stream>>>(x, xh, xl, n4);
  }
  // wgu: per expert (2I x H) -> (H x 2I)
  k_transpose_split<<<dim3(H_ / 32, O_ / 32, E_), dim3(32, 8), 0, stream>>>(
      wgu, guh, gul, O_, H_);
  // wd: per expert (H x I) -> (I x H)
  k_transpose_split<<<dim3(I_ / 32, H_ / 32, E_), dim3(32, 8), 0, stream>>>(
      wd, wdh, wdl, H_, I_);

  // Pass 1: fused gate/up GEMM + SiLU -> act (hi/lo f16)
  k_gemm1<<<dim3(I_ / 32, T_ / 64, E_), dim3(32), 0, stream>>>(
      xh, xl, guh, gul, acth, actl);

  // Pass 2: down-projection GEMM -> fp32 output
  k_gemm2<<<dim3(H_ / 32, T_ / 64, E_), dim3(32), 0, stream>>>(
      acth, actl, wdh, wdl, out);
}
